// MoMQA_76802605187651
// MI455X (gfx1250) — matrix operand order repaired and checked by reading, not compiled, not run on hardware
//
#include <hip/hip_runtime.h>

#define N_EMBD   2048
#define N_HEAD   16
#define HEAD_DIM 128
#define BATCH    2
#define T_SEQ    2048
#define N_TOK    (BATCH * T_SEQ)   // 4096

typedef __attribute__((ext_vector_type(16))) __bf16 v16bf;
typedef __attribute__((ext_vector_type(8)))  __bf16 v8bf;
typedef __attribute__((ext_vector_type(4)))  __bf16 v4bf;
typedef __attribute__((ext_vector_type(8)))  float  v8f;

__device__ __forceinline__ __bf16 f2bf(float f) {
    unsigned u = __builtin_bit_cast(unsigned, f);
    u += 0x7FFFu + ((u >> 16) & 1u);                 // round-to-nearest-even
    unsigned short s = (unsigned short)(u >> 16);
    return __builtin_bit_cast(__bf16, s);
}

__device__ __forceinline__ v16bf cat16(v8bf a, v8bf b) {
    return __builtin_shufflevector(a, b, 0,1,2,3,4,5,6,7,8,9,10,11,12,13,14,15);
}

// ---------------------------------------------------------------------------
// fp32 -> bf16 elementwise convert, 4 elements / thread
// ---------------------------------------------------------------------------
__global__ void cvt_bf16_kernel(const float* __restrict__ in, __bf16* __restrict__ out, int n4) {
    int i = blockIdx.x * blockDim.x + threadIdx.x;
    if (i < n4) {
        const float4 v = *(const float4*)(in + (size_t)i * 4);
        v4bf o;
        o[0] = f2bf(v.x); o[1] = f2bf(v.y); o[2] = f2bf(v.z); o[3] = f2bf(v.w);
        *(v4bf*)(out + (size_t)i * 4) = o;
    }
}

// ---------------------------------------------------------------------------
// Gate: logits = x @ W_gate^T ; idx = argmax softmax ; top1 = max softmax.
// One wave per token. Lane l: head = l&15, half = l>>4 handles 1024 of K.
// ---------------------------------------------------------------------------
__global__ void __launch_bounds__(32)
gate_kernel(const float* __restrict__ x, const float* __restrict__ Wg,
            int* __restrict__ idx, float* __restrict__ top1) {
    const int m    = blockIdx.x;
    const int lane = threadIdx.x;
    const int h    = lane & 15;
    const int half = lane >> 4;
    const float* xr = x  + (size_t)m * N_EMBD + half * (N_EMBD / 2);
    const float* wr = Wg + (size_t)h * N_EMBD + half * (N_EMBD / 2);
    float acc = 0.0f;
    for (int c = 0; c < N_EMBD / 2; c += 4) {
        const float4 xv = *(const float4*)(xr + c);
        const float4 wv = *(const float4*)(wr + c);
        acc += xv.x * wv.x + xv.y * wv.y + xv.z * wv.z + xv.w * wv.w;
    }
    acc += __shfl_xor(acc, 16, 32);          // fold halves: every lane has full logit(h)
    float v  = acc;
    int   vi = h;
    #pragma unroll
    for (int msk = 8; msk >= 1; msk >>= 1) { // max + argmax (first index on ties)
        float ov = __shfl_xor(v,  msk, 32);
        int   oi = __shfl_xor(vi, msk, 32);
        if (ov > v || (ov == v && oi < vi)) { v = ov; vi = oi; }
    }
    float e = __expf(acc - v);
    #pragma unroll
    for (int msk = 8; msk >= 1; msk >>= 1) e += __shfl_xor(e, msk, 32);
    if (lane == 0) { idx[m] = vi; top1[m] = 1.0f / e; }   // softmax max = 1/sum(exp(l-lmax))
}

// ---------------------------------------------------------------------------
// Q projection: q = x @ W_q^T via bf16 WMMA, written bf16 as [B][H][T][hd].
// Wave tile: M=64, N=64 -> 4 A-frags x 4 B-frags = 16 wmma per K-step of 32.
// Each B-frag (weights) is reused by 4 A-frags: ~32 flop per L2 byte.
// ---------------------------------------------------------------------------
__global__ void __launch_bounds__(32)
qproj_kernel(const __bf16* __restrict__ xb, const __bf16* __restrict__ wqb,
             __bf16* __restrict__ qb) {
    const int lane = threadIdx.x;
    const int row  = lane & 15;
    const int hi   = lane >> 4;
    const int n0   = blockIdx.x * 64;
    const int m0   = blockIdx.y * 64;

    v8f acc[4][4] = {};
    const __bf16* xr[4];
    #pragma unroll
    for (int mf = 0; mf < 4; ++mf)
        xr[mf] = xb + (size_t)(m0 + mf * 16 + row) * N_EMBD;

    for (int k0 = 0; k0 < N_EMBD; k0 += 32) {
        const int ka = k0 + hi * 8;     // A layout: half-wave K {0..7,16..23}/{8..15,24..31}
        v16bf a[4];
        #pragma unroll
        for (int mf = 0; mf < 4; ++mf)
            a[mf] = cat16(*(const v8bf*)(xr[mf] + ka), *(const v8bf*)(xr[mf] + ka + 16));
        #pragma unroll
        for (int nf = 0; nf < 4; ++nf) {
            // B layout: lane col = row; half-wave K {0..7,16..23}/{8..15,24..31}
            const __bf16* wr = wqb + (size_t)(n0 + nf * 16 + row) * N_EMBD + k0 + hi * 8;
            v16bf bf = cat16(*(const v8bf*)wr, *(const v8bf*)(wr + 16));
            #pragma unroll
            for (int mf = 0; mf < 4; ++mf)
                acc[mf][nf] = __builtin_amdgcn_wmma_f32_16x16x32_bf16(false, a[mf], false, bf,
                                                                      (short)0, acc[mf][nf],
                                                                      false, false);
        }
    }
    // D layout element (M = r + 8*hi, N = row); scatter into qb[B][H][T][hd]
    #pragma unroll
    for (int mf = 0; mf < 4; ++mf) {
        #pragma unroll
        for (int nf = 0; nf < 4; ++nf) {
            #pragma unroll
            for (int r = 0; r < 8; ++r) {
                const int mg = m0 + mf * 16 + hi * 8 + r;
                const int ng = n0 + nf * 16 + row;
                const int b = mg >> 11, t = mg & (T_SEQ - 1);
                const int h = ng >> 7,  d = ng & (HEAD_DIM - 1);
                qb[((size_t)(b * N_HEAD + h) * T_SEQ + t) * HEAD_DIM + d] = f2bf(acc[mf][nf][r]);
            }
        }
    }
}

// ---------------------------------------------------------------------------
// K/V gather-GEMM: only the selected head per token.
//   k_shared[t,d] = x[t,:] . W_qkv[C + idx*128 + d, :]
//   v_sharedT[d,t] = top1 * (x[t,:] . W_qkv[2C + idx*128 + d, :])   (stored [B][hd][T])
// Block = token, 128 threads (one per d), x row staged in LDS.
// ---------------------------------------------------------------------------
__global__ void __launch_bounds__(128)
kv_gather_kernel(const float* __restrict__ x, const float* __restrict__ Wqkv,
                 const int* __restrict__ idx, const float* __restrict__ top1,
                 __bf16* __restrict__ kb, __bf16* __restrict__ vbT) {
    __shared__ float xs[N_EMBD];
    const int m = blockIdx.x;
    const int d = threadIdx.x;
    const int b = m >> 11, t = m & (T_SEQ - 1);
    const float* xr = x + (size_t)m * N_EMBD;
    for (int c = d; c < N_EMBD; c += 128) xs[c] = xr[c];
    __syncthreads();
    const int   hs = idx[m];
    const float g  = top1[m];
    const float* wk = Wqkv + (size_t)(N_EMBD     + hs * HEAD_DIM + d) * N_EMBD;
    const float* wv = Wqkv + (size_t)(2 * N_EMBD + hs * HEAD_DIM + d) * N_EMBD;
    float ak = 0.0f, av = 0.0f;
    for (int c = 0; c < N_EMBD; c += 4) {
        const float4 xv = *(const float4*)(xs + c);
        const float4 kv = *(const float4*)(wk + c);
        const float4 vv = *(const float4*)(wv + c);
        ak += xv.x * kv.x + xv.y * kv.y + xv.z * kv.z + xv.w * kv.w;
        av += xv.x * vv.x + xv.y * vv.y + xv.z * vv.z + xv.w * vv.w;
    }
    kb [(size_t)m * HEAD_DIM + d]               = f2bf(ak);
    vbT[((size_t)b * HEAD_DIM + d) * T_SEQ + t] = f2bf(g * av);
}

// ---------------------------------------------------------------------------
// Flash attention: one wave per (b, h, 16-query tile); 32-key steps, causal.
// 8 wmma for S=Q·K^T + 8 wmma for O+=P·V per step. P routed D->A via LDS.
// Next step's K/V cachelines prefetched (global_prefetch_b8) across softmax.
// ---------------------------------------------------------------------------
__global__ void __launch_bounds__(32)
attn_kernel(const __bf16* __restrict__ qb, const __bf16* __restrict__ kb,
            const __bf16* __restrict__ vbT, float* __restrict__ out) {
    __shared__ __bf16 plds[16 * 32];
    const int lane = threadIdx.x;
    const int row  = lane & 15;
    const int hi   = lane >> 4;
    const int q0   = blockIdx.x * 16;
    const int bh   = blockIdx.y;          // b*16 + h
    const int b    = bh >> 4;
    const int h    = bh & 15;

    // Q tile: four 16x32 A-fragments across hd=128
    v16bf qa[4];
    {
        const __bf16* qr = qb + ((size_t)bh * T_SEQ + (q0 + row)) * HEAD_DIM;
        #pragma unroll
        for (int f = 0; f < 4; ++f) {
            const int d0 = f * 32 + hi * 8;
            qa[f] = cat16(*(const v8bf*)(qr + d0), *(const v8bf*)(qr + d0 + 16));
        }
    }

    v8f   o[8] = {};
    float mrow[8], lrow[8];
    #pragma unroll
    for (int r = 0; r < 8; ++r) { mrow[r] = -3.0e38f; lrow[r] = 0.0f; }

    const float scale = 0.08838834764831843f;   // 1/sqrt(128)
    const int   klast = (q0 + 15) & ~31;
    const __bf16* kbb  = kb  + (size_t)b * T_SEQ * HEAD_DIM;
    const __bf16* vbb  = vbT + (size_t)b * HEAD_DIM * T_SEQ;

    for (int kt0 = 0; kt0 <= klast; kt0 += 32) {
        // ---- S = Q K^T (16 x 32) ----
        v8f s[2] = {};
        #pragma unroll
        for (int f = 0; f < 4; ++f) {
            #pragma unroll
            for (int nf = 0; nf < 2; ++nf) {
                const __bf16* kr = kbb + (size_t)(kt0 + nf * 16 + row) * HEAD_DIM
                                       + f * 32 + hi * 8;
                v16bf kf = cat16(*(const v8bf*)kr, *(const v8bf*)(kr + 16));
                s[nf] = __builtin_amdgcn_wmma_f32_16x16x32_bf16(false, qa[f], false, kf,
                                                                (short)0, s[nf], false, false);
            }
        }
        // prefetch next step's K rows (32 rows, one per lane) and V row segments
        if (kt0 + 32 <= klast) {
            __builtin_prefetch(kbb + (size_t)(kt0 + 32 + lane) * HEAD_DIM, 0, 0);
            __builtin_prefetch(vbb + (size_t)(lane * 4) * T_SEQ + kt0 + 32, 0, 0);
        }
        // ---- causal mask + online softmax (lane owns rows hi*8 .. hi*8+7) ----
        #pragma unroll
        for (int r = 0; r < 8; ++r) {
            const int qi = q0 + hi * 8 + r;
            float s0 = s[0][r] * scale;
            float s1 = s[1][r] * scale;
            if (kt0 + row      > qi) s0 = -3.0e38f;
            if (kt0 + 16 + row > qi) s1 = -3.0e38f;
            float mx = fmaxf(s0, s1);
            #pragma unroll
            for (int msk = 8; msk >= 1; msk >>= 1) mx = fmaxf(mx, __shfl_xor(mx, msk, 32));
            const float mnew = fmaxf(mrow[r], mx);
            const float corr = __expf(mrow[r] - mnew);
            const float p0   = __expf(s0 - mnew);
            const float p1   = __expf(s1 - mnew);
            float ps = p0 + p1;
            #pragma unroll
            for (int msk = 8; msk >= 1; msk >>= 1) ps += __shfl_xor(ps, msk, 32);
            lrow[r] = lrow[r] * corr + ps;
            mrow[r] = mnew;
            #pragma unroll
            for (int nf = 0; nf < 8; ++nf) o[nf][r] *= corr;
            plds[(hi * 8 + r) * 32 + row]      = f2bf(p0);   // (M, kloc=row)
            plds[(hi * 8 + r) * 32 + 16 + row] = f2bf(p1);   // (M, kloc=16+row)
        }
        asm volatile("s_wait_dscnt 0x0" ::: "memory");        // LDS stores -> loads (same wave)
        // ---- P as 16x32 A-fragment ----
        const __bf16* pr = &plds[row * 32 + hi * 8];
        const v16bf pa = cat16(*(const v8bf*)pr, *(const v8bf*)(pr + 16));
        // ---- O += P V (all 128 dims) ----
        #pragma unroll
        for (int nf = 0; nf < 8; ++nf) {
            const __bf16* vr = vbb + (size_t)(nf * 16 + row) * T_SEQ + kt0 + hi * 8;
            v16bf vf = cat16(*(const v8bf*)vr, *(const v8bf*)(vr + 16));
            o[nf] = __builtin_amdgcn_wmma_f32_16x16x32_bf16(false, pa, false, vf,
                                                            (short)0, o[nf], false, false);
        }
    }
    // ---- normalize + store [B,T,C] fp32 ----
    float inv[8];
    #pragma unroll
    for (int r = 0; r < 8; ++r) inv[r] = 1.0f / lrow[r];
    #pragma unroll
    for (int nf = 0; nf < 8; ++nf) {
        #pragma unroll
        for (int r = 0; r < 8; ++r) {
            const int t = q0 + hi * 8 + r;
            const int d = nf * 16 + row;
            out[((size_t)b * T_SEQ + t) * N_EMBD + h * HEAD_DIM + d] = o[nf][r] * inv[r];
        }
    }
}

// ---------------------------------------------------------------------------
extern "C" void kernel_launch(void* const* d_in, const int* in_sizes, int n_in,
                              void* d_out, int out_size, void* d_ws, size_t ws_size,
                              hipStream_t stream) {
    (void)in_sizes; (void)n_in; (void)out_size; (void)ws_size;
    const float* x    = (const float*)d_in[0];   // [B,T,C]
    const float* Wg   = (const float*)d_in[1];   // [H,C]
    const float* Wqkv = (const float*)d_in[2];   // [3C,C]
    float* out = (float*)d_out;                  // [B,T,C]

    char* ws = (char*)d_ws;                      // ~44 MB total
    __bf16* xb   = (__bf16*)(ws);                              // x bf16          16 MB
    __bf16* wqb  = (__bf16*)(ws + (16u << 20));                // W_q bf16         8 MB
    __bf16* qb   = (__bf16*)(ws + (24u << 20));                // q [B,H,T,hd]    16 MB
    __bf16* kb   = (__bf16*)(ws + (40u << 20));                // k_shared [B,T,hd] 1 MB
    __bf16* vbT  = (__bf16*)(ws + (41u << 20));                // v_sharedT [B,hd,T] 1 MB
    int*    idxp = (int*)  (ws + (42u << 20));                 // [4096]
    float*  top1 = (float*)(ws + (42u << 20) + (1u << 16));    // [4096]

    const int nx4 = (N_TOK * N_EMBD) / 4;    // 2097152
    const int nw4 = (N_EMBD * N_EMBD) / 4;   // 1048576 (q rows only)
    cvt_bf16_kernel<<<(nx4 + 255) / 256, 256, 0, stream>>>(x, xb, nx4);
    cvt_bf16_kernel<<<(nw4 + 255) / 256, 256, 0, stream>>>(Wqkv, wqb, nw4);
    gate_kernel<<<N_TOK, 32, 0, stream>>>(x, Wg, idxp, top1);
    qproj_kernel<<<dim3(N_EMBD / 64, N_TOK / 64), 32, 0, stream>>>(xb, wqb, qb);
    kv_gather_kernel<<<N_TOK, 128, 0, stream>>>(x, Wqkv, idxp, top1, kb, vbT);
    attn_kernel<<<dim3(T_SEQ / 16, BATCH * N_HEAD), 32, 0, stream>>>(qb, kb, vbT, out);
}
